// EncoderDecoder_64579128262667
// MI455X (gfx1250) — matrix operand order repaired and checked by reading, not compiled, not run on hardware
//
#include <hip/hip_runtime.h>
#include <hip/hip_bf16.h>

#define VV   32000
#define HH   512
#define H3   1536
#define TT   128
#define BB   64
#define ML   32
#define BOS_TOK 2

typedef __attribute__((ext_vector_type(16))) _Float16 v16h;
typedef __attribute__((ext_vector_type(8)))  _Float16 v8h;
typedef __attribute__((ext_vector_type(8)))  float    v8f;

// ---------------- f32 -> f16 convert ----------------
__global__ void f32_to_f16_k(const float* __restrict__ src, _Float16* __restrict__ dst, int n) {
    int i = blockIdx.x * blockDim.x + threadIdx.x;
    if (i < n) dst[i] = (_Float16)src[i];
}

// ---------------- init h=0, tok=BOS ----------------
__global__ void init_state_k(float* __restrict__ h, _Float16* __restrict__ h_h, int* __restrict__ tok) {
    int i = blockIdx.x * blockDim.x + threadIdx.x;
    if (i < BB * HH) { h[i] = 0.0f; h_h[i] = (_Float16)0.0f; }
    if (i < BB) tok[i] = BOS_TOK;
}

// ---- shared fragment helpers (A/B k order: contiguous k = 16*h + s) ----
__device__ __forceinline__ v16h load_afrag(const _Float16* __restrict__ arow, int k0, int akoff) {
    v8h lo = *(const v8h*)(arow + k0 + akoff);
    v8h hi = *(const v8h*)(arow + k0 + akoff + 8);
    v16h a = {};
    #pragma unroll
    for (int i = 0; i < 8; ++i) { a[i] = lo[i]; a[i + 8] = hi[i]; }
    return a;
}

__device__ __forceinline__ void store_ctile(float* __restrict__ D, const v8f& c,
                                            int m0, int n0, int N, int lh, int l, float bv) {
    float* drow = D + (size_t)(m0 + lh * 8) * N + (n0 + l);
    #pragma unroll
    for (int r = 0; r < 8; ++r)
        drow[(size_t)r * N] = c[r] + bv;
}

// ---------------- latency-optimized WMMA GEMM: one wave per 16x16 tile ----------------
// Used for the tiny sequential recurrence GEMMs (max wave parallelism per step).
__global__ __launch_bounds__(256) void wmma_gemm_f16_k(
    const _Float16* __restrict__ A,
    const int*      __restrict__ a_idx,
    const _Float16* __restrict__ table,
    const _Float16* __restrict__ W,
    const float*    __restrict__ bias,
    float*          __restrict__ D,
    int M, int N, int K)
{
    const int wave = threadIdx.x >> 5;
    const int lane = threadIdx.x & 31;
    const int tid  = blockIdx.x * 8 + wave;
    const int tiles_n = N >> 4;
    const int tiles_m = M >> 4;
    if (tid >= tiles_m * tiles_n) return;          // wave-uniform: EXEC stays all-1s
    const int m0 = (tid / tiles_n) << 4;
    const int n0 = (tid % tiles_n) << 4;
    const int lh = lane >> 4;
    const int l  = lane & 15;
    const int akoff = lh * 16;

    const int ar = m0 + l;
    const _Float16* arow = a_idx ? (table + (size_t)a_idx[ar] * K)
                                 : (A     + (size_t)ar       * K);
    const _Float16* brow = W + (size_t)(n0 + l) * K + lh * 16;

    v8f c = {};
    for (int k0 = 0; k0 < K; k0 += 32) {
        v16h a = load_afrag(arow, k0, akoff);
        v16h b = *(const v16h*)(brow + k0);
        __builtin_prefetch(brow + k0 + 64, 0, 0);
        c = __builtin_amdgcn_wmma_f32_16x16x32_f16(false, a, false, b, (short)0, c, false, false);
    }
    const float bv = bias ? bias[n0 + l] : 0.0f;
    store_ctile(D, c, m0, n0, N, lh, l, bv);
}

// ---------------- throughput-optimized WMMA GEMM: one wave per 64-row M-strip x 16 N ----------------
// 4 WMMAs per B-fragment load => 4x less W traffic from L2. Requires M % 64 == 0.
__global__ __launch_bounds__(256) void wmma_gemm_f16_strip4_k(
    const _Float16* __restrict__ A,
    const int*      __restrict__ a_idx,
    const _Float16* __restrict__ table,
    const _Float16* __restrict__ W,
    const float*    __restrict__ bias,
    float*          __restrict__ D,
    int M, int N, int K)
{
    const int wave = threadIdx.x >> 5;
    const int lane = threadIdx.x & 31;
    const int tid  = blockIdx.x * 8 + wave;
    const int tiles_n  = N >> 4;
    const int strips_m = M >> 6;
    if (tid >= strips_m * tiles_n) return;         // wave-uniform
    const int m0 = (tid / tiles_n) << 6;
    const int n0 = (tid % tiles_n) << 4;
    const int lh = lane >> 4;
    const int l  = lane & 15;
    const int akoff = lh * 16;

    const _Float16* arow0;
    const _Float16* arow1;
    const _Float16* arow2;
    const _Float16* arow3;
    if (a_idx) {
        arow0 = table + (size_t)a_idx[m0      + l] * K;
        arow1 = table + (size_t)a_idx[m0 + 16 + l] * K;
        arow2 = table + (size_t)a_idx[m0 + 32 + l] * K;
        arow3 = table + (size_t)a_idx[m0 + 48 + l] * K;
    } else {
        arow0 = A + (size_t)(m0      + l) * K;
        arow1 = A + (size_t)(m0 + 16 + l) * K;
        arow2 = A + (size_t)(m0 + 32 + l) * K;
        arow3 = A + (size_t)(m0 + 48 + l) * K;
    }
    const _Float16* brow = W + (size_t)(n0 + l) * K + lh * 16;

    v8f c0 = {}, c1 = {}, c2 = {}, c3 = {};
    for (int k0 = 0; k0 < K; k0 += 32) {
        v16h b = *(const v16h*)(brow + k0);
        __builtin_prefetch(brow + k0 + 64, 0, 0);
        v16h a0 = load_afrag(arow0, k0, akoff);
        v16h a1 = load_afrag(arow1, k0, akoff);
        v16h a2 = load_afrag(arow2, k0, akoff);
        v16h a3 = load_afrag(arow3, k0, akoff);
        c0 = __builtin_amdgcn_wmma_f32_16x16x32_f16(false, a0, false, b, (short)0, c0, false, false);
        c1 = __builtin_amdgcn_wmma_f32_16x16x32_f16(false, a1, false, b, (short)0, c1, false, false);
        c2 = __builtin_amdgcn_wmma_f32_16x16x32_f16(false, a2, false, b, (short)0, c2, false, false);
        c3 = __builtin_amdgcn_wmma_f32_16x16x32_f16(false, a3, false, b, (short)0, c3, false, false);
    }
    const float bv = bias ? bias[n0 + l] : 0.0f;
    store_ctile(D, c0, m0,      n0, N, lh, l, bv);
    store_ctile(D, c1, m0 + 16, n0, N, lh, l, bv);
    store_ctile(D, c2, m0 + 32, n0, N, lh, l, bv);
    store_ctile(D, c3, m0 + 48, n0, N, lh, l, bv);
}

// ---------------- GRU gate update (PyTorch order r,z,n); optional length mask ----------------
__global__ void gru_update_k(const float* __restrict__ gi, const float* __restrict__ gh,
                             float* __restrict__ h, _Float16* __restrict__ h_h,
                             const int* __restrict__ lengths, int t)
{
    int i = blockIdx.x * blockDim.x + threadIdx.x;
    if (i >= BB * HH) return;
    const int b = i / HH;
    const int j = i - b * HH;
    const size_t rb = (size_t)b * H3;
    const float ir = gi[rb + j],          hr = gh[rb + j];
    const float iz = gi[rb + HH + j],     hz = gh[rb + HH + j];
    const float in_ = gi[rb + 2*HH + j],  hn = gh[rb + 2*HH + j];
    const float r = 1.0f / (1.0f + expf(-(ir + hr)));
    const float z = 1.0f / (1.0f + expf(-(iz + hz)));
    const float n = tanhf(in_ + r * hn);
    const float hv = h[i];
    float hnew = (1.0f - z) * n + z * hv;
    if (lengths && !(t < lengths[b])) hnew = hv;   // packed-seq freeze
    h[i]   = hnew;
    h_h[i] = (_Float16)hnew;
}

// ---------------- per-row argmax over V columns (first max wins, like jnp.argmax) ----------------
__global__ void argmax_rows_k(const float* __restrict__ logits, int* __restrict__ tok) {
    const int b = blockIdx.x;
    const float* row = logits + (size_t)b * VV;
    float best = -__builtin_huge_valf();
    int   bidx = 0x7fffffff;
    for (int j = threadIdx.x; j < VV; j += blockDim.x) {
        float v = row[j];
        if (v > best || (v == best && j < bidx)) { best = v; bidx = j; }
    }
    __shared__ float sv[256];
    __shared__ int   si[256];
    sv[threadIdx.x] = best; si[threadIdx.x] = bidx;
    __syncthreads();
    for (int s = 128; s > 0; s >>= 1) {
        if (threadIdx.x < s) {
            float ov = sv[threadIdx.x + s]; int oi = si[threadIdx.x + s];
            if (ov > sv[threadIdx.x] || (ov == sv[threadIdx.x] && oi < si[threadIdx.x])) {
                sv[threadIdx.x] = ov; si[threadIdx.x] = oi;
            }
        }
        __syncthreads();
    }
    if (threadIdx.x == 0) tok[b] = si[0];
}

static inline size_t align_up(size_t x) { return (x + 255) & ~(size_t)255; }

extern "C" void kernel_launch(void* const* d_in, const int* in_sizes, int n_in,
                              void* d_out, int out_size, void* d_ws, size_t ws_size,
                              hipStream_t stream) {
    (void)in_sizes; (void)n_in; (void)out_size; (void)ws_size;
    const int*   batch_X = (const int*)  d_in[0];   // (T,B) int32
    const int*   lengths = (const int*)  d_in[1];   // (B,)
    const float* emb_enc = (const float*)d_in[3];
    const float* Wih_e   = (const float*)d_in[4];
    const float* Whh_e   = (const float*)d_in[5];
    const float* bih_e   = (const float*)d_in[6];
    const float* bhh_e   = (const float*)d_in[7];
    const float* emb_dec = (const float*)d_in[8];
    const float* Wih_d   = (const float*)d_in[9];
    const float* Whh_d   = (const float*)d_in[10];
    const float* bih_d   = (const float*)d_in[11];
    const float* bhh_d   = (const float*)d_in[12];
    const float* Wout    = (const float*)d_in[13];
    const float* bout    = (const float*)d_in[14];
    float* out = (float*)d_out;

    // ---- workspace carve-up ----
    char* ws = (char*)d_ws; size_t off = 0;
    auto take = [&](size_t bytes) { char* p = ws + off; off = align_up(off + bytes); return p; };
    _Float16* emb_enc_h = (_Float16*)take((size_t)VV * HH * 2);
    _Float16* Wih_e_h   = (_Float16*)take((size_t)H3 * HH * 2);
    _Float16* Whh_e_h   = (_Float16*)take((size_t)H3 * HH * 2);
    _Float16* emb_dec_h = (_Float16*)take((size_t)VV * HH * 2);
    _Float16* Wih_d_h   = (_Float16*)take((size_t)H3 * HH * 2);
    _Float16* Whh_d_h   = (_Float16*)take((size_t)H3 * HH * 2);
    _Float16* Wout_h    = (_Float16*)take((size_t)VV * HH * 2);
    float*    gi_all    = (float*)   take((size_t)TT * BB * H3 * 4);
    float*    gh_buf    = (float*)   take((size_t)BB * H3 * 4);
    float*    gi_d      = (float*)   take((size_t)BB * H3 * 4);
    float*    h         = (float*)   take((size_t)BB * HH * 4);
    _Float16* h_h       = (_Float16*)take((size_t)BB * HH * 2);
    int*      tok       = (int*)     take((size_t)BB * 4);

    // ---- one-time f16 conversions ----
    auto conv = [&](const float* s, _Float16* d, int n) {
        f32_to_f16_k<<<(n + 255) / 256, 256, 0, stream>>>(s, d, n);
    };
    conv(emb_enc, emb_enc_h, VV * HH);
    conv(Wih_e,   Wih_e_h,   H3 * HH);
    conv(Whh_e,   Whh_e_h,   H3 * HH);
    conv(emb_dec, emb_dec_h, VV * HH);
    conv(Wih_d,   Wih_d_h,   H3 * HH);
    conv(Whh_d,   Whh_d_h,   H3 * HH);
    conv(Wout,    Wout_h,    VV * HH);

    init_state_k<<<(BB * HH + 255) / 256, 256, 0, stream>>>(h, h_h, tok);

    // ---- encoder input-side GEMM hoisted out of the recurrence: (T*B) x 3H (throughput kernel) ----
    {
        int waves = ((TT * BB) / 64) * (H3 / 16);        // 128 strips * 96 n-tiles = 12288
        wmma_gemm_f16_strip4_k<<<waves / 8, 256, 0, stream>>>(
            nullptr, batch_X, emb_enc_h, Wih_e_h, bih_e, gi_all, TT * BB, H3, HH);
    }

    // ---- encoder recurrence (latency kernel: max wave parallelism per step) ----
    const int gh_blocks = ((BB / 16) * (H3 / 16)) / 8;   // 384/8 = 48
    for (int t = 0; t < TT; ++t) {
        wmma_gemm_f16_k<<<gh_blocks, 256, 0, stream>>>(
            h_h, nullptr, nullptr, Whh_e_h, bhh_e, gh_buf, BB, H3, HH);
        gru_update_k<<<(BB * HH + 255) / 256, 256, 0, stream>>>(
            gi_all + (size_t)t * BB * H3, gh_buf, h, h_h, lengths, t);
    }

    // ---- greedy decoder ----
    const int lg_blocks = ((BB / 64) * (VV / 16)) / 8;   // 2000 waves / 8 = 250 blocks (strip4)
    for (int s = 0; s < ML; ++s) {
        wmma_gemm_f16_k<<<gh_blocks, 256, 0, stream>>>(       // gi = emb_dec[tok] @ Wih_d^T + bih_d
            nullptr, tok, emb_dec_h, Wih_d_h, bih_d, gi_d, BB, H3, HH);
        wmma_gemm_f16_k<<<gh_blocks, 256, 0, stream>>>(       // gh = h @ Whh_d^T + bhh_d
            h_h, nullptr, nullptr, Whh_d_h, bhh_d, gh_buf, BB, H3, HH);
        gru_update_k<<<(BB * HH + 255) / 256, 256, 0, stream>>>(
            gi_d, gh_buf, h, h_h, nullptr, 0);
        float* logits = out + (size_t)s * BB * VV;
        wmma_gemm_f16_strip4_k<<<lg_blocks, 256, 0, stream>>>( // logits = h @ Wout^T + bout
            h_h, nullptr, nullptr, Wout_h, bout, logits, BB, VV, HH);
        argmax_rows_k<<<BB, 256, 0, stream>>>(logits, tok);
    }
}
